// HybridQNN_27290222198803
// MI455X (gfx1250) — matrix operand order repaired and checked by reading, not compiled, not run on hardware
//
#include <hip/hip_runtime.h>
#include <hip/hip_bf16.h>

#define BATCH   16384
#define IN_FEAT 256
#define HIDDEN  128
#define NQ      10
#define NL      4

typedef __attribute__((ext_vector_type(16))) __bf16 v16bf;
typedef __attribute__((ext_vector_type(8)))  float  v8f;
typedef __attribute__((ext_vector_type(4)))  unsigned int u32x4;
typedef __attribute__((ext_vector_type(8)))  int    i32x8;
typedef __attribute__((ext_vector_type(4)))  int    i32x4;

// ---------------------------------------------------------------------------
// fp32 -> bf16 conversion
// ---------------------------------------------------------------------------
__global__ void cvt_bf16(const float* __restrict__ in, __bf16* __restrict__ out, int n) {
    int i = blockIdx.x * blockDim.x + threadIdx.x;
    if (i < n) out[i] = (__bf16)in[i];
}

// ---------------------------------------------------------------------------
// TDM: async 2-D bf16 tile load, global -> LDS (rows packed contiguously).
// D# encoding per CDNA5 ISA:
//  group0: [1:0]=count(1), [63:32]=lds_addr, [120:64]=global_addr, [127:126]=2
//  group1: [17:16]=data_size(1 => 2B), [79:48]=tensor_dim0, [111:80]=tensor_dim1,
//          [127:112]=tile_dim0, [143:128]=tile_dim1, [207:160]=tensor_dim0_stride
// groups 2/3 zero (2-D tile, tile_dim2=0).
// ---------------------------------------------------------------------------
__device__ __forceinline__ void tdm_load_2d_bf16(
    unsigned lds_off, const __bf16* gptr,
    unsigned tile_cols, unsigned tile_rows,
    unsigned tensor_cols, unsigned tensor_rows, unsigned row_stride_elems)
{
    unsigned long long ga = (unsigned long long)(uintptr_t)gptr;
    u32x4 g0;
    g0[0] = 1u;                                            // count=1, user mode
    g0[1] = lds_off;                                       // LDS byte address
    g0[2] = (unsigned)ga;                                  // global_addr[31:0]
    g0[3] = (unsigned)((ga >> 32) & 0x1FFFFFFu) | (2u << 30); // addr[56:32] | type=2
    i32x8 g1;
    g1[0] = (int)(1u << 16);                               // data_size=1 (2 bytes)
    g1[1] = (int)((tensor_cols & 0xFFFFu) << 16);          // tensor_dim0 lo16
    g1[2] = (int)((tensor_cols >> 16) | ((tensor_rows & 0xFFFFu) << 16));
    g1[3] = (int)((tensor_rows >> 16) | (tile_cols << 16)); // tensor_dim1 hi | tile_dim0
    g1[4] = (int)(tile_rows & 0xFFFFu);                    // tile_dim1 (tile_dim2=0)
    g1[5] = (int)row_stride_elems;                         // tensor_dim0_stride lo32
    g1[6] = 0;                                             // stride hi, dim1_stride lo
    g1[7] = 0;
    i32x4 z4 = {};
#if __clang_major__ >= 23
    i32x8 z8 = {};
    __builtin_amdgcn_tensor_load_to_lds(g0, g1, z4, z4, z8, 0);
#else
    __builtin_amdgcn_tensor_load_to_lds(g0, g1, z4, z4, 0);
#endif
}

// ---------------------------------------------------------------------------
// GEMM1: h[16384,128] = relu( x[16384,256] @ W1[128,256]^T + b1 )
// bf16 WMMA, TDM double-buffered LDS staging, vectorized ds_load fragments.
// Block = 256 threads (8 waves); M-tile 128, N = 128; K loop of 8 x 32.
// ---------------------------------------------------------------------------
__global__ __launch_bounds__(256) void gemm1_wmma(
    const __bf16* __restrict__ xb,   // [16384][256]
    const __bf16* __restrict__ w1b,  // [128][256]
    const float*  __restrict__ b1,   // [128]
    float*        __restrict__ h)    // [16384][128]
{
    __shared__ __align__(128) __bf16 As[2][128][32];  // x rows   [m][k]
    __shared__ __align__(128) __bf16 Bs[2][128][32];  // W1 rows  [n][k]

    const int tid  = threadIdx.x;
    const int lane = tid & 31;
    const int wv   = tid >> 5;          // 0..7
    const int m0   = blockIdx.x * 128;  // row block
    const int r    = lane & 15;
    const int half = lane >> 4;

    v8f acc[8] = {};

    if (wv == 0) {
        tdm_load_2d_bf16((unsigned)(uintptr_t)&As[0][0][0],
                         xb + (size_t)m0 * IN_FEAT, 32, 128, IN_FEAT, BATCH, IN_FEAT);
        tdm_load_2d_bf16((unsigned)(uintptr_t)&Bs[0][0][0],
                         w1b, 32, 128, IN_FEAT, HIDDEN, IN_FEAT);
    }

    for (int kt = 0; kt < 8; ++kt) {
        const int cur = kt & 1;
        if (wv == 0) __builtin_amdgcn_s_wait_tensorcnt(0);
        __syncthreads();   // TDM data visible; prev iter reads of other buf done

        if (wv == 0 && kt < 7) {   // prefetch next K-step into the other buffer
            const int K0n = (kt + 1) * 32;
            tdm_load_2d_bf16((unsigned)(uintptr_t)&As[cur ^ 1][0][0],
                             xb + (size_t)m0 * IN_FEAT + K0n, 32, 128, IN_FEAT, BATCH, IN_FEAT);
            tdm_load_2d_bf16((unsigned)(uintptr_t)&Bs[cur ^ 1][0][0],
                             w1b + K0n, 32, 128, IN_FEAT, HIDDEN, IN_FEAT);
        }

        // A fragment (16x32): element e -> K = 16*(e/8) + 8*half + e%8
        // => two contiguous 16B runs per lane.
        v16bf a;
        *(uint4*)&a       = *(const uint4*)&As[cur][wv * 16 + r][half * 8];
        *((uint4*)&a + 1) = *(const uint4*)&As[cur][wv * 16 + r][16 + half * 8];

        #pragma unroll
        for (int nt = 0; nt < 8; ++nt) {
            // B fragment (32x16): element e -> K = 16*(e/8) + 8*half + e%8, col n = nt*16 + r
            // => two contiguous 16B runs per lane from Bs[n][k].
            v16bf bfr;
            *(uint4*)&bfr       = *(const uint4*)&Bs[cur][nt * 16 + r][half * 8];
            *((uint4*)&bfr + 1) = *(const uint4*)&Bs[cur][nt * 16 + r][16 + half * 8];
            acc[nt] = __builtin_amdgcn_wmma_f32_16x16x32_bf16(
                false, a, false, bfr, (short)0, acc[nt], false, false);
        }
    }

    // epilogue: C layout lane<16: M=v,N=lane ; lane>=16: M=v+8,N=lane-16
    #pragma unroll
    for (int nt = 0; nt < 8; ++nt) {
        int col = nt * 16 + r;
        float bias = b1[col];
        #pragma unroll
        for (int v = 0; v < 8; ++v) {
            int row = m0 + wv * 16 + half * 8 + v;
            h[(size_t)row * HIDDEN + col] = fmaxf(acc[nt][v] + bias, 0.0f);
        }
    }
}

// ---------------------------------------------------------------------------
// GEMM2 + tanh: angles[b][w] = tanh(b2[w] + sum_k h[b][k]*W2[w][k])
// ---------------------------------------------------------------------------
__global__ void gemm2_tanh(const float* __restrict__ h, const float* __restrict__ W2,
                           const float* __restrict__ b2, float* __restrict__ ang)
{
    int idx = blockIdx.x * blockDim.x + threadIdx.x;      // BATCH*NQ
    if (idx >= BATCH * NQ) return;
    int b = idx / NQ, w = idx % NQ;
    float acc = b2[w];
    const float* hr = h + (size_t)b * HIDDEN;
    const float* wr = W2 + (size_t)w * HIDDEN;
    #pragma unroll 8
    for (int k = 0; k < HIDDEN; ++k) acc = fmaf(hr[k], wr[k], acc);
    ang[idx] = tanhf(acc);
}

// ---------------------------------------------------------------------------
// Quantum circuit: one wave32 per batch element. State = 1024 complex amps:
// amplitude index i = reg*32 + lane; wire w <-> bit (9-w).
// wires 0..4 -> register bits, wires 5..9 -> lane bits (shfl_xor).
// ---------------------------------------------------------------------------
__global__ __launch_bounds__(256) void qcircuit(
    const float* __restrict__ ang,   // [BATCH][10]
    const float* __restrict__ qw,    // [4][10][3]
    float*       __restrict__ q)     // [BATCH][10]
{
    const int lane = threadIdx.x & 31;
    const int b = blockIdx.x * (blockDim.x >> 5) + (threadIdx.x >> 5);

    // --- initial RY layer on |0..0> is a product state ---
    float cw[NQ], sw[NQ];
    #pragma unroll
    for (int w = 0; w < NQ; ++w) {
        float th = 0.5f * ang[(size_t)b * NQ + w];
        sincosf(th, &sw[w], &cw[w]);
    }
    float lp = 1.0f;
    #pragma unroll
    for (int w = 5; w < NQ; ++w) lp *= ((lane >> (9 - w)) & 1) ? sw[w] : cw[w];

    float sr[32], si[32];
    #pragma unroll
    for (int r = 0; r < 32; ++r) {
        float rp = 1.0f;
        #pragma unroll
        for (int w = 0; w < 5; ++w) rp *= ((r >> (4 - w)) & 1) ? sw[w] : cw[w];
        sr[r] = rp * lp;
        si[r] = 0.0f;
    }

    // --- entangling layers ---
    #pragma unroll
    for (int l = 0; l < NL; ++l) {
        // Rot gates
        #pragma unroll
        for (int w = 0; w < NQ; ++w) {
            const float phi = qw[(l * NQ + w) * 3 + 0];
            const float th  = qw[(l * NQ + w) * 3 + 1];
            const float om  = qw[(l * NQ + w) * 3 + 2];
            float c, s, sa, ca, sb, cb;
            sincosf(0.5f * th, &s, &c);
            sincosf(0.5f * (phi + om), &sa, &ca);
            sincosf(0.5f * (phi - om), &sb, &cb);
            const float u00r =  c * ca, u00i = -c * sa;
            const float u01r = -s * cb, u01i = -s * sb;
            const float u10r =  s * cb, u10i = -s * sb;
            const float u11r =  c * ca, u11i =  c * sa;
            const int p = 9 - w;
            if (p >= 5) {                       // register-bit gate
                const int tb = 1 << (p - 5);
                #pragma unroll
                for (int r = 0; r < 32; ++r) if (!(r & tb)) {
                    const int r1 = r | tb;
                    float a0r = sr[r], a0i = si[r], a1r = sr[r1], a1i = si[r1];
                    sr[r]  = u00r*a0r - u00i*a0i + u01r*a1r - u01i*a1i;
                    si[r]  = u00r*a0i + u00i*a0r + u01r*a1i + u01i*a1r;
                    sr[r1] = u10r*a0r - u10i*a0i + u11r*a1r - u11i*a1i;
                    si[r1] = u10r*a0i + u10i*a0r + u11r*a1i + u11i*a1r;
                }
            } else {                            // lane-bit gate (cross-lane)
                const int lm = 1 << p;
                const bool hi = (lane & lm) != 0;
                const float csr = hi ? u11r : u00r, csi = hi ? u11i : u00i;
                const float cor = hi ? u10r : u01r, coi = hi ? u10i : u01i;
                #pragma unroll
                for (int r = 0; r < 32; ++r) {
                    float pr = __shfl_xor(sr[r], lm, 32);
                    float pi = __shfl_xor(si[r], lm, 32);
                    float ar = sr[r], ai = si[r];
                    sr[r] = csr*ar - csi*ai + cor*pr - coi*pi;
                    si[r] = csr*ai + csi*ar + cor*pi + coi*pr;
                }
            }
        }
        // CNOT ring: control i, target (i+off)%10
        const int off = (l % 9) + 1;
        #pragma unroll
        for (int i = 0; i < NQ; ++i) {
            const int c = i, t = (i + off) % NQ;
            const int pc = 9 - c, pt = 9 - t;
            if (pc >= 5 && pt >= 5) {            // both reg bits: swap regs
                const int cb2 = 1 << (pc - 5), tb = 1 << (pt - 5);
                #pragma unroll
                for (int r = 0; r < 32; ++r) if ((r & cb2) && !(r & tb)) {
                    const int r1 = r | tb;
                    float tr = sr[r]; sr[r] = sr[r1]; sr[r1] = tr;
                    float ti = si[r]; si[r] = si[r1]; si[r1] = ti;
                }
            } else if (pc >= 5) {                // ctrl reg, target lane: shfl regs
                const int cb2 = 1 << (pc - 5), tm = 1 << pt;
                #pragma unroll
                for (int r = 0; r < 32; ++r) if (r & cb2) {
                    sr[r] = __shfl_xor(sr[r], tm, 32);
                    si[r] = __shfl_xor(si[r], tm, 32);
                }
            } else if (pt >= 5) {                // ctrl lane, target reg: cond swap
                const int tb = 1 << (pt - 5);
                const bool ctl = ((lane >> pc) & 1) != 0;
                #pragma unroll
                for (int r = 0; r < 32; ++r) if (!(r & tb)) {
                    const int r1 = r | tb;
                    float ar = sr[r], ai = si[r], br = sr[r1], bi = si[r1];
                    sr[r]  = ctl ? br : ar;  si[r]  = ctl ? bi : ai;
                    sr[r1] = ctl ? ar : br;  si[r1] = ctl ? ai : bi;
                }
            } else {                             // both lane bits
                const int cm = 1 << pc, tm = 1 << pt;
                const bool ctl = (lane & cm) != 0;
                #pragma unroll
                for (int r = 0; r < 32; ++r) {
                    float orr = __shfl_xor(sr[r], tm, 32);
                    float oii = __shfl_xor(si[r], tm, 32);
                    sr[r] = ctl ? orr : sr[r];
                    si[r] = ctl ? oii : si[r];
                }
            }
        }
    }

    // --- measurement: q[b][w] = sum_i |amp_i|^2 * (1 - 2*bit_{9-w}(i)) ---
    float T = 0.0f, S[5] = {0, 0, 0, 0, 0};
    #pragma unroll
    for (int r = 0; r < 32; ++r) {
        float p = sr[r] * sr[r] + si[r] * si[r];
        T += p;
        #pragma unroll
        for (int w = 0; w < 5; ++w) S[w] += ((r >> (4 - w)) & 1) ? -p : p;
    }
    float outv[NQ];
    #pragma unroll
    for (int w = 0; w < 5; ++w) outv[w] = S[w];
    #pragma unroll
    for (int w = 5; w < NQ; ++w) outv[w] = ((lane >> (9 - w)) & 1) ? -T : T;
    #pragma unroll
    for (int w = 0; w < NQ; ++w) {
        float v = outv[w];
        #pragma unroll
        for (int offm = 16; offm >= 1; offm >>= 1) v += __shfl_xor(v, offm, 32);
        outv[w] = v;
    }
    if (lane == 0) {
        #pragma unroll
        for (int w = 0; w < NQ; ++w) q[(size_t)b * NQ + w] = outv[w];
    }
}

// ---------------------------------------------------------------------------
// Head: out[b] = b4 + sum_h W4[h]*relu(b3[h] + sum_w W3[h][w]*q[b][w])
// ---------------------------------------------------------------------------
__global__ void head(const float* __restrict__ q, const float* __restrict__ W3,
                     const float* __restrict__ b3, const float* __restrict__ W4,
                     const float* __restrict__ b4, float* __restrict__ out)
{
    int b = blockIdx.x * blockDim.x + threadIdx.x;
    if (b >= BATCH) return;
    float qv[NQ];
    #pragma unroll
    for (int w = 0; w < NQ; ++w) qv[w] = q[(size_t)b * NQ + w];
    float acc = b4[0];
    for (int hh = 0; hh < HIDDEN; ++hh) {
        float d = b3[hh];
        #pragma unroll
        for (int w = 0; w < NQ; ++w) d = fmaf(W3[hh * NQ + w], qv[w], d);
        acc = fmaf(W4[hh], fmaxf(d, 0.0f), acc);
    }
    out[b] = acc;
}

// ---------------------------------------------------------------------------
extern "C" void kernel_launch(void* const* d_in, const int* in_sizes, int n_in,
                              void* d_out, int out_size, void* d_ws, size_t ws_size,
                              hipStream_t stream)
{
    const float* x  = (const float*)d_in[0];
    const float* W1 = (const float*)d_in[1];
    const float* b1 = (const float*)d_in[2];
    const float* W2 = (const float*)d_in[3];
    const float* b2 = (const float*)d_in[4];
    const float* qw = (const float*)d_in[5];
    const float* W3 = (const float*)d_in[6];
    const float* b3 = (const float*)d_in[7];
    const float* W4 = (const float*)d_in[8];
    const float* b4 = (const float*)d_in[9];
    float* out = (float*)d_out;

    char* ws = (char*)d_ws;
    __bf16* xb  = (__bf16*)(ws);                       //  8 MB
    __bf16* w1b = (__bf16*)(ws + 8388608);             // 64 KB
    float*  h   = (float*)(ws + 8454144);              //  8 MB
    float*  ang = (float*)(ws + 16842752);             // 640 KB
    float*  q   = (float*)(ws + 17498112);             // 640 KB

    cvt_bf16<<<(BATCH * IN_FEAT + 255) / 256, 256, 0, stream>>>(x, xb, BATCH * IN_FEAT);
    cvt_bf16<<<(HIDDEN * IN_FEAT + 255) / 256, 256, 0, stream>>>(W1, w1b, HIDDEN * IN_FEAT);
    gemm1_wmma<<<BATCH / 128, 256, 0, stream>>>(xb, w1b, b1, h);
    gemm2_tanh<<<(BATCH * NQ + 255) / 256, 256, 0, stream>>>(h, W2, b2, ang);
    qcircuit<<<BATCH / 8, 256, 0, stream>>>(ang, qw, q);
    head<<<(BATCH + 255) / 256, 256, 0, stream>>>(q, W3, b3, W4, b4, out);
}
